// GraphAttention_2946347565084
// MI455X (gfx1250) — matrix operand order repaired and checked by reading, not compiled, not run on hardware
//
#include <hip/hip_runtime.h>

typedef __attribute__((ext_vector_type(16))) _Float16 v16h;
typedef __attribute__((ext_vector_type(8)))  _Float16 v8h;
typedef __attribute__((ext_vector_type(4)))  _Float16 v4h;
typedef __attribute__((ext_vector_type(8)))  float    v8f;

#define ALPHA   0.2f
#define NS      32      // nsample
#define DD      64      // feature dim D
#define KDIM    67      // C + D (permuted: K0..63 = delta_h, K64..66 = delta_p)
#define KPAD    96      // K padded to 3 x 32
#define WAVES   2       // waves per block (one point per wave per iteration)
#define GSTRIDE 66      // gf row stride: 8B-aligned stores, conflict-free column reads
#define NBLOCKS 4096    // persistent blocks, grid-stride over points

__global__ __launch_bounds__(WAVES * 32)
void graph_attn_wmma_kernel(const float* __restrict__ center_xyz,
                            const float* __restrict__ center_feature,
                            const float* __restrict__ grouped_xyz,
                            const float* __restrict__ grouped_feature,
                            const float* __restrict__ amat,
                            float* __restrict__ out, int P)
{
    // a pre-swizzled into B-fragment layout: [ki][ni][lane][e], 16 contiguous f16/lane
    __shared__ __align__(16) _Float16 a_frag_lds[3 * 4 * 32 * 16];   // 12 KB
    __shared__ __align__(16) _Float16 x_lds[WAVES][NS * KPAD];       // 6 KB / wave
    __shared__ __align__(16) float    gf_lds[WAVES][NS * GSTRIDE];   // 8.25 KB / wave

    const int tid  = threadIdx.x;
    const int lane = tid & 31;
    const int wv   = tid >> 5;
    const int l15  = lane & 15;
    const int hi   = (lane >> 4) & 1;

    // ---- Stage a (67x64 f32) into f16 B-fragment layout, K-permuted + zero-padded ----
    // B k order used here (same as A): element e -> K = ki*32 + (lane>=16 ? 16:0) + e, N = ni*16 + (lane&15)
    for (int f = tid; f < 3 * 4 * 32 * 16; f += blockDim.x) {
        int e  = f & 15;
        int l  = (f >> 4) & 31;
        int ni = (f >> 9) & 3;
        int ki = f >> 11;
        int K  = ki * 32 + ((l & 16) ? 16 : 0) + e;
        int N  = ni * 16 + (l & 15);
        float v = 0.0f;
        if (K < KDIM) {
            int kr = (K < 64) ? (K + 3) : (K - 64);   // K-permutation: delta_h first
            v = amat[kr * DD + N];
        }
        a_frag_lds[f] = (_Float16)v;
    }
    __syncthreads();

    // ---- B fragments: load once, hold in registers for all points ----
    v16h bf[3][4];
    #pragma unroll
    for (int ki = 0; ki < 3; ++ki) {
        #pragma unroll
        for (int ni = 0; ni < 4; ++ni) {
            const v8h* pb = (const v8h*)(a_frag_lds + ((ki * 4 + ni) * 32 + lane) * 16);
            v8h b0 = pb[0];
            v8h b1 = pb[1];
            bf[ki][ni] = __builtin_shufflevector(b0, b1,
                0, 1, 2, 3, 4, 5, 6, 7, 8, 9, 10, 11, 12, 13, 14, 15);
        }
    }

    _Float16* xw  = x_lds[wv];
    float*    gfw = gf_lds[wv];

    const int pstride = gridDim.x * WAVES;
    for (int p = blockIdx.x * WAVES + wv; p < P; p += pstride) {
        const float*  gx  = grouped_xyz + (size_t)p * (NS * 3);
        const float4* gf4 = (const float4*)(grouped_feature + (size_t)p * (NS * DD));
        const float*  cx  = center_xyz + (size_t)p * 3;
        const float4* cf4 = (const float4*)(center_feature + (size_t)p * DD);

        // prefetch next point's grouped_feature tile (8 KB spread over 32 lanes)
        {
            int pn = p + pstride;
            if (pn < P) {
                const char* nf = (const char*)(grouped_feature + (size_t)pn * (NS * DD));
                __builtin_prefetch(nf + lane * 256, 0, 1);
            }
        }

        // ---- stage x (f16, K-permuted) and grouped_feature (f32) into LDS ----
        #pragma unroll
        for (int i = 0; i < 16; ++i) {        // 32 rows * 16 float4 / 32 lanes
            int idx = i * 32 + lane;
            int s = idx >> 4;
            int c = idx & 15;                  // == lane & 15 (constant per lane)
            float4 g  = gf4[idx];
            float4 cc = cf4[c];
            int gb = s * GSTRIDE + c * 4;
            float2 glo; glo.x = g.x; glo.y = g.y;
            float2 ghi; ghi.x = g.z; ghi.y = g.w;
            *(float2*)(gfw + gb)     = glo;    // 8B-aligned b64 stores
            *(float2*)(gfw + gb + 2) = ghi;
            v4h dh = { (_Float16)(cc.x - g.x), (_Float16)(cc.y - g.y),
                       (_Float16)(cc.z - g.z), (_Float16)(cc.w - g.w) };
            *(v4h*)(xw + s * KPAD + c * 4) = dh;          // K = 4c..4c+3 (delta_h)
        }
        {
            float c0 = cx[0], c1 = cx[1], c2 = cx[2];
            int s = lane;                                  // lane == sample row
            v4h dp = { (_Float16)(c0 - gx[s * 3 + 0]),
                       (_Float16)(c1 - gx[s * 3 + 1]),
                       (_Float16)(c2 - gx[s * 3 + 2]),
                       (_Float16)0.0f };
            *(v4h*)(xw + s * KPAD + 64) = dp;              // K 64..67 (67 = pad)
            const v4h z4 = { (_Float16)0.0f, (_Float16)0.0f,
                             (_Float16)0.0f, (_Float16)0.0f };
            #pragma unroll
            for (int j = 0; j < 7; ++j)                    // K 68..95 = pad
                *(v4h*)(xw + s * KPAD + 68 + j * 4) = z4;
        }
        // same-wave LDS RAW: DS pipeline is in-order; compiler inserts s_wait_dscnt

        // ---- e = leaky_relu(x @ a) via v_wmma_f32_16x16x32_f16 ----
        v8f acc[2][4];
        const v8f vzero = {0.f, 0.f, 0.f, 0.f, 0.f, 0.f, 0.f, 0.f};
        #pragma unroll
        for (int mi = 0; mi < 2; ++mi)
            #pragma unroll
            for (int ni = 0; ni < 4; ++ni)
                acc[mi][ni] = vzero;

        #pragma unroll
        for (int ki = 0; ki < 3; ++ki) {
            // A order used here (same as B): one contiguous 16-element run per lane (K = kbase + hi*16 + e)
            v16h af[2];
            #pragma unroll
            for (int mi = 0; mi < 2; ++mi) {
                const v8h* pa = (const v8h*)(xw + (mi * 16 + l15) * KPAD + ki * 32 + hi * 16);
                v8h a0 = pa[0];       // K = kbase + hi*16 .. +7
                v8h a1 = pa[1];       // K = kbase + hi*16 + 8 .. +15
                af[mi] = __builtin_shufflevector(a0, a1,
                    0, 1, 2, 3, 4, 5, 6, 7, 8, 9, 10, 11, 12, 13, 14, 15);
            }
            #pragma unroll
            for (int ni = 0; ni < 4; ++ni) {
                #pragma unroll
                for (int mi = 0; mi < 2; ++mi) {
                    acc[mi][ni] = __builtin_amdgcn_wmma_f32_16x16x32_f16(
                        false, af[mi], false, bf[ki][ni],
                        (short)0, acc[mi][ni], false, false);
                }
            }
        }

        // ---- LeakyReLU in-register: alpha<1 => leaky(v) = max(v, alpha*v) ----
        #pragma unroll
        for (int mi = 0; mi < 2; ++mi)
            #pragma unroll
            for (int ni = 0; ni < 4; ++ni)
                #pragma unroll
                for (int r = 0; r < 8; ++r) {
                    float v = acc[mi][ni][r];
                    acc[mi][ni][r] = fmaxf(v, ALPHA * v);
                }

        // ---- register-resident softmax over s + weighted pooling ----
        // Column d = ni*16+l15 lives on lane pair (l15, l15+16):
        //   this half owns 16 rows s = mi*16 + r + hi*8; combine halves via shfl_xor(16).
        float resv[4];
        #pragma unroll
        for (int ni = 0; ni < 4; ++ni) {
            const int dcol = ni * 16 + l15;
            // max over this half's 16 logits, then cross-half combine
            float m = acc[0][ni][0];
            #pragma unroll
            for (int mi = 0; mi < 2; ++mi)
                #pragma unroll
                for (int r = 0; r < 8; ++r)
                    m = fmaxf(m, acc[mi][ni][r]);
            m = fmaxf(m, __shfl_xor(m, 16, 32));
            // exp + partial sum + partial weighted sum (gf column read: conflict-free)
            float psum = 0.0f, pw = 0.0f;
            #pragma unroll
            for (int mi = 0; mi < 2; ++mi) {
                #pragma unroll
                for (int r = 0; r < 8; ++r) {
                    int s = mi * 16 + r + hi * 8;
                    float ex = __expf(acc[mi][ni][r] - m);
                    psum += ex;
                    pw   += ex * gfw[s * GSTRIDE + dcol];
                }
            }
            psum += __shfl_xor(psum, 16, 32);
            pw   += __shfl_xor(pw,   16, 32);
            resv[ni] = pw / psum;
        }

        // both halves now hold all 4 column results; split stores: low half d<32, high half d>=32
        float* outp = out + (size_t)p * DD;
        float r0 = hi ? resv[2] : resv[0];
        float r1 = hi ? resv[3] : resv[1];
        outp[hi * 32 + l15]      = r0;
        outp[hi * 32 + 16 + l15] = r1;
    }
}

extern "C" void kernel_launch(void* const* d_in, const int* in_sizes, int n_in,
                              void* d_out, int out_size, void* d_ws, size_t ws_size,
                              hipStream_t stream) {
    const float* center_xyz      = (const float*)d_in[0];
    const float* center_feature  = (const float*)d_in[1];
    const float* grouped_xyz     = (const float*)d_in[2];
    const float* grouped_feature = (const float*)d_in[3];
    const float* amat            = (const float*)d_in[4];
    float* out = (float*)d_out;

    int P = in_sizes[0] / 3;                          // B * npoint
    int blocks = (P + WAVES - 1) / WAVES;
    if (blocks > NBLOCKS) blocks = NBLOCKS;           // persistent, grid-stride
    graph_attn_wmma_kernel<<<blocks, WAVES * 32, 0, stream>>>(
        center_xyz, center_feature, grouped_xyz, grouped_feature, amat, out, P);
}
